// SynthesisBlock_78357383348541
// MI455X (gfx1250) — matrix operand order repaired and checked by reading, not compiled, not run on hardware
//
#include <hip/hip_runtime.h>

// ============================================================================
// StyleGAN2 SynthesisBlock for MI455X (gfx1250, wave32, WMMA + async-LDS DMA).
//
// Math refactor: modulated conv with shared const input
//   y_b = d_b[o] * conv3x3(conv_w, s_b[c] * const[c])
// so the heavy op is ONE shared-weight implicit GEMM per batch sample:
//   M = 512 (out ch), N = 4096 (pixels), K = 512*9, done in f16 WMMA
//   (v_wmma_f32_16x16x32_f16) with f32 accumulate.
//
// gfx1250 paths used:
//   * v_wmma_f32_16x16x32_f16 (8 per k-step, software-pipelined by compiler)
//   * global_load_async_to_lds_b128 for the ping-pong A-tile refill
//     (ASYNCcnt-tracked DMA overlapping the WMMA burst; s_wait_asynccnt 0
//     before the tap barrier)
//   * global_prefetch hints for the next channel chunk
//
// Scratch (d_ws) requirement: ~5.9 MB
//   q[512*512] s[16*512] s2[16*512] d[16*512] (f32), wT[9*512*512] (f16)
// ============================================================================

typedef _Float16 f16;
typedef _Float16 v16h __attribute__((ext_vector_type(16)));
typedef float    v8f  __attribute__((ext_vector_type(8)));

#define NPIX  4096
#define BATCH 16

// scratch float offsets
#define Q_FOFF   0
#define S_FOFF   (512*512)
#define S2_FOFF  (S_FOFF  + 16*512)
#define D_FOFF   (S2_FOFF + 16*512)
#define WT_FOFF  (D_FOFF  + 16*512)   // f16 payload starts here

// ---- async global->LDS support (guarded; sync fallback keeps compile OK) ---
#if defined(__HIP_DEVICE_COMPILE__) && \
    __has_builtin(__builtin_amdgcn_global_load_async_to_lds_b128)
#define HAVE_ASYNC 1
#else
#define HAVE_ASYNC 0
#endif

#if HAVE_ASYNC
// The builtin expects: (int __vector_size(16) __device__ *src,
//                       lds ptr, imm offset, imm cpol)
typedef int v4i_ __attribute__((vector_size(16)));
#define AS1C(p) ((__attribute__((address_space(1))) v4i_*)(p))
#define AS3P(p) ((__attribute__((address_space(3))) v4i_*)(p))
#if __has_builtin(__builtin_amdgcn_s_wait_asynccnt)
#define WAIT_ASYNC() __builtin_amdgcn_s_wait_asynccnt(0)
#else
#define WAIT_ASYNC() asm volatile("s_wait_asynccnt 0x0" ::: "memory")
#endif
#endif

// ---------------------------------------------------------------------------
// q[o,c] = sum_tap conv_w[o,c,tap]^2
__global__ __launch_bounds__(256) void k_q(const float* __restrict__ cw,
                                           float* __restrict__ q) {
  int g = blockIdx.x * 256 + threadIdx.x;          // 512*512 exact
  const float* p = cw + (size_t)g * 9;
  float a = 0.f;
#pragma unroll
  for (int t = 0; t < 9; ++t) a += p[t] * p[t];
  q[g] = a;
}

// ---------------------------------------------------------------------------
// styles: s[b,c] = ws[b,0]·conv_aff_w[c] + conv_aff_b[c]
//         s2[b,c] = (ws[b,1]·rgb_aff_w[c] + rgb_aff_b[c]) / sqrt(512)
__global__ __launch_bounds__(256) void k_styles(const float* __restrict__ ws,
                                                const float* __restrict__ caw,
                                                const float* __restrict__ cab,
                                                const float* __restrict__ raw,
                                                const float* __restrict__ rab,
                                                float* __restrict__ sOut,
                                                float* __restrict__ s2Out) {
  int g = blockIdx.x * 256 + threadIdx.x;          // 2*16*512 exact
  int slot = g >> 13;
  int b = (g >> 9) & 15;
  int c = g & 511;
  const float* wv = ws + ((size_t)b * 2 + slot) * 512;
  const float* aw = (slot ? raw : caw) + (size_t)c * 512;
  float acc = 0.f;
  for (int j = 0; j < 512; ++j) acc += wv[j] * aw[j];
  if (slot == 0) sOut[b * 512 + c] = acc + cab[c];
  else           s2Out[b * 512 + c] = (acc + rab[c]) * 0.04419417382415922f;
}

// ---------------------------------------------------------------------------
// d[b,o] = rsqrt(sum_c s[b,c]^2 * q[o,c] + 1e-8)
__global__ __launch_bounds__(256) void k_dcoef(const float* __restrict__ s,
                                               const float* __restrict__ q,
                                               float* __restrict__ d) {
  int g = blockIdx.x * 256 + threadIdx.x;          // 16*512 exact
  int b = g >> 9, o = g & 511;
  const float* sb = s + (size_t)b * 512;
  const float* qo = q + (size_t)o * 512;
  float acc = 1e-8f;
  for (int c = 0; c < 512; ++c) { float t = sb[c]; acc += t * t * qo[c]; }
  d[g] = rsqrtf(acc);
}

// ---------------------------------------------------------------------------
// wT[tap][o][c] = (f16) conv_w[o][c][tap]   (coalesced A-tile source)
__global__ __launch_bounds__(256) void k_wt(const float* __restrict__ cw,
                                            f16* __restrict__ wT) {
  int g = blockIdx.x * 256 + threadIdx.x;          // 9*512*512 exact
  int c = g & 511;
  int o = (g >> 9) & 511;
  int tap = g >> 18;
  wT[g] = (f16)cw[((size_t)o * 512 + c) * 9 + tap];
}

// ---------------------------------------------------------------------------
// Main implicit-GEMM modulated conv + noise + bias + lrelu*sqrt(2).
// Block: 256 threads = 8 waves (2x4). Tile: 128 out-ch x 128 pixels.
// K loop: 16 channel chunks (32 ch) x 9 taps; ping-pong A tiles refilled by
// async global->LDS b128 DMA overlapping the WMMA burst.
#define APAD 40                       // 80B rows: 16B-aligned b128 segments
#define XCOLS 66
#define XPAD 36

__global__ __launch_bounds__(256) void k_modconv(
    const float* __restrict__ cst,   // const [512,64,64]
    const f16*   __restrict__ wT,    // [9][512][512] f16
    const float* __restrict__ cb,    // conv bias [512]
    const float* __restrict__ nz,    // noise_const [4096]
    const float* __restrict__ nsp,   // noise strength [1]
    const float* __restrict__ sBuf,  // styles [16,512]
    const float* __restrict__ dBuf,  // demod [16,512]
    float* __restrict__ xOut)        // [16,512,4096]
{
  __shared__ f16 aT[2 * 128 * APAD];          // ping-pong A: [buf][m][k]
  __shared__ f16 xp[4 * XCOLS * XPAD];        // patch: [row][col][c] (c-contig)
  __shared__ float dBlk[128];
  __shared__ float bBlk[128];

  const int tid  = threadIdx.x;
  const int lane = tid & 31;
  const int wv   = tid >> 5;
  const int wm   = wv >> 2;                   // 0..1 (64 rows each)
  const int wn   = wv & 3;                    // 0..3 (32 cols each)
  const int b    = blockIdx.z;
  const int o0   = blockIdx.y * 128;
  const int pixBase = blockIdx.x * 128;       // 2 image rows
  const int y0   = pixBase >> 6;

  if (tid < 128) {
    dBlk[tid] = dBuf[b * 512 + o0 + tid];
    bBlk[tid] = cb[o0 + tid];
  }

  v8f acc[4][2];
#pragma unroll
  for (int i = 0; i < 4; ++i)
#pragma unroll
    for (int j = 0; j < 2; ++j)
      acc[i][j] = (v8f){0.f, 0.f, 0.f, 0.f, 0.f, 0.f, 0.f, 0.f};

  const int kb    = (lane >> 4) * 16;         // A half-select
  const int khalf = (lane >> 4) * 16;         // B half-select
  int buf = 0;

  for (int ck = 0; ck < 16; ++ck) {
    const int c0 = ck * 32;

    // ---- stage modulated, zero-padded input patch (32ch x 4rows x 66cols) --
    for (int i = tid; i < 32 * 4 * XCOLS; i += 256) {
      int c   = i / (4 * XCOLS);              // channel-major: coalesced rows
      int rem = i % (4 * XCOLS);
      int row = rem / XCOLS;
      int col = rem % XCOLS;
      int gy = y0 - 1 + row;
      int gx = col - 1;
      float v = 0.f;
      if ((unsigned)gy < 64u && (unsigned)gx < 64u)
        v = cst[((size_t)(c0 + c)) * NPIX + gy * 64 + gx] *
            sBuf[b * 512 + c0 + c];
      xp[(row * XCOLS + col) * XPAD + c] = (f16)v;
    }
    if (ck == 0) {                            // prime A tile (tap 0)
#if HAVE_ASYNC
      {
        const f16* src = wT + (size_t)o0 * 512 + c0;
#pragma unroll
        for (int t = 0; t < 2; ++t) {         // 512 x 16B segs, 2 per thread
          int sIdx = tid + 256 * t;
          int m = sIdx >> 2, sg = sIdx & 3;
          __builtin_amdgcn_global_load_async_to_lds_b128(
              AS1C(src + (size_t)m * 512 + sg * 8),
              AS3P(aT + m * APAD + sg * 8), 0, 0);
        }
        WAIT_ASYNC();
      }
#else
      for (int i = tid; i < 128 * 32; i += 256) {
        int k = i & 31, m = i >> 5;
        aT[m * APAD + k] = wT[(size_t)(o0 + m) * 512 + c0 + k];
      }
#endif
    }
    if (ck < 15)                              // hint next chunk into caches
      __builtin_prefetch(cst + ((size_t)(c0 + 32)) * NPIX, 0, 1);
    __syncthreads();

    for (int tap = 0; tap < 9; ++tap) {
      // ---- refill the other A buffer while this one feeds the WMMAs -------
      {
        int ntap = tap + 1, nc0 = c0;
        bool doLoad = true;
        if (ntap == 9) { ntap = 0; nc0 = c0 + 32; doLoad = (ck < 15); }
        if (doLoad) {
          const f16* src = wT + ((size_t)ntap * 512 + o0) * 512 + nc0;
          f16* dst = aT + (buf ^ 1) * 128 * APAD;
#if HAVE_ASYNC
#pragma unroll
          for (int t = 0; t < 2; ++t) {       // async DMA, overlaps compute
            int sIdx = tid + 256 * t;
            int m = sIdx >> 2, sg = sIdx & 3;
            __builtin_amdgcn_global_load_async_to_lds_b128(
                AS1C(src + (size_t)m * 512 + sg * 8),
                AS3P(dst + m * APAD + sg * 8), 0, 0);
          }
#else
          for (int i = tid; i < 128 * 32; i += 256) {
            int k = i & 31, m = i >> 5;
            dst[m * APAD + k] = src[(size_t)m * 512 + k];
          }
#endif
        }
      }

      const int dy = tap / 3, dx = tap % 3;
      union FragU { v16h v; unsigned long long u[4]; };
      FragU af[4], bfr[2];

      // A fragments: row m, contiguous K[kb..kb+15]
      const f16* abase = aT + buf * 128 * APAD;
#pragma unroll
      for (int i = 0; i < 4; ++i) {
        const f16* p = abase + (64 * wm + 16 * i + (lane & 15)) * APAD;
        af[i].u[0] = *(const unsigned long long*)(p + kb);
        af[i].u[1] = *(const unsigned long long*)(p + kb + 4);
        af[i].u[2] = *(const unsigned long long*)(p + kb + 8);
        af[i].u[3] = *(const unsigned long long*)(p + kb + 12);
      }
      // B fragments: pixel col, contiguous K-run of 16 channels
#pragma unroll
      for (int j = 0; j < 2; ++j) {
        int cl = 32 * wn + 16 * j + (lane & 15);
        int r2 = (cl >> 6) + dy;
        int xc = (cl & 63) + dx;
        const f16* p = &xp[(r2 * XCOLS + xc) * XPAD + khalf];
        bfr[j].u[0] = ((const unsigned long long*)p)[0];
        bfr[j].u[1] = ((const unsigned long long*)p)[1];
        bfr[j].u[2] = ((const unsigned long long*)p)[2];
        bfr[j].u[3] = ((const unsigned long long*)p)[3];
      }

#pragma unroll
      for (int i = 0; i < 4; ++i)
#pragma unroll
        for (int j = 0; j < 2; ++j)
          acc[i][j] = __builtin_amdgcn_wmma_f32_16x16x32_f16(
              false, af[i].v, false, bfr[j].v, (short)0, acc[i][j],
              false, false);

#if HAVE_ASYNC
      WAIT_ASYNC();                           // own DMA done before barrier
#endif
      __syncthreads();
      buf ^= 1;
    }
  }

  // ---- epilogue: demod, noise, bias, lrelu * sqrt(2) ----------------------
  const float ns = nsp[0];
#pragma unroll
  for (int i = 0; i < 4; ++i) {
#pragma unroll
    for (int j = 0; j < 2; ++j) {
      int nl = 32 * wn + 16 * j + (lane & 15);
      int p  = pixBase + nl;
      float noisev = nz[p] * ns;
      int mbase = 64 * wm + 16 * i + ((lane >> 4) << 3);
#pragma unroll
      for (int r = 0; r < 8; ++r) {
        int m = mbase + r;
        float y = acc[i][j][r] * dBlk[m] + noisev + bBlk[m];
        y = (y > 0.f ? y : 0.2f * y) * 1.4142135623730951f;
        xOut[((size_t)b * 512 + (o0 + m)) * NPIX + p] = y;
      }
    }
  }
}

// ---------------------------------------------------------------------------
// ToRGB: img[b,o,p] = sum_c x[b,c,p] * rgb_w[o,c] * s2[b,c] + rgb_b[o]
__global__ __launch_bounds__(256) void k_torgb(const float* __restrict__ x,
                                               const float* __restrict__ rw,
                                               const float* __restrict__ rb,
                                               const float* __restrict__ s2,
                                               float* __restrict__ img) {
  __shared__ float co[3 * 512];
  int b = blockIdx.y;
  int p = blockIdx.x * 256 + threadIdx.x;
  for (int i = threadIdx.x; i < 3 * 512; i += 256) {
    int o = i >> 9, c = i & 511;
    co[i] = rw[o * 512 + c] * s2[b * 512 + c];
  }
  __syncthreads();
  float a0 = 0.f, a1 = 0.f, a2 = 0.f;
  const float* xb = x + (size_t)b * 512 * NPIX + p;
  for (int c = 0; c < 512; ++c) {
    float v = xb[(size_t)c * NPIX];
    a0 += co[c] * v;
    a1 += co[512 + c] * v;
    a2 += co[1024 + c] * v;
  }
  img[((size_t)b * 3 + 0) * NPIX + p] = a0 + rb[0];
  img[((size_t)b * 3 + 1) * NPIX + p] = a1 + rb[1];
  img[((size_t)b * 3 + 2) * NPIX + p] = a2 + rb[2];
}

// ===========================================================================
extern "C" void kernel_launch(void* const* d_in, const int* in_sizes, int n_in,
                              void* d_out, int out_size, void* d_ws,
                              size_t ws_size, hipStream_t stream) {
  const float* ws  = (const float*)d_in[0];   // [16,2,512]
  const float* cst = (const float*)d_in[1];   // [512,64,64]
  const float* cw  = (const float*)d_in[2];   // [512,512,3,3]
  const float* cb  = (const float*)d_in[3];   // [512]
  const float* caw = (const float*)d_in[4];   // [512,512]
  const float* cab = (const float*)d_in[5];   // [512]
  const float* nz  = (const float*)d_in[6];   // [64,64]
  const float* nst = (const float*)d_in[7];   // [1]
  const float* rw  = (const float*)d_in[8];   // [3,512,1,1]
  const float* rb  = (const float*)d_in[9];   // [3]
  const float* raw = (const float*)d_in[10];  // [512,512]
  const float* rab = (const float*)d_in[11];  // [512]

  float* out  = (float*)d_out;
  float* xOut = out;                              // [16,512,64,64]
  float* img  = out + (size_t)BATCH * 512 * NPIX; // [16,3,64,64]

  float* wsf   = (float*)d_ws;
  float* q     = wsf + Q_FOFF;
  float* s     = wsf + S_FOFF;
  float* s2    = wsf + S2_FOFF;
  float* dcoef = wsf + D_FOFF;
  f16*   wT    = (f16*)(wsf + WT_FOFF);

  k_q     <<<1024, 256, 0, stream>>>(cw, q);
  k_styles<<<  64, 256, 0, stream>>>(ws, caw, cab, raw, rab, s, s2);
  k_dcoef <<<  32, 256, 0, stream>>>(s, q, dcoef);
  k_wt    <<<9216, 256, 0, stream>>>(cw, wT);

  dim3 g(32, 4, 16);                 // pixel tiles x out-ch tiles x batch
  k_modconv<<<g, 256, 0, stream>>>(cst, wT, cb, nz, nst, s, dcoef, xOut);

  dim3 g2(16, 16);                   // pixel blocks x batch
  k_torgb<<<g2, 256, 0, stream>>>(xOut, rw, rb, s2, img);
}
